// Decoder_32779190403498
// MI455X (gfx1250) — compile-verified, matrix operand order checked by reading, not run on hardware
//
#include <hip/hip_runtime.h>
#include <hip/hip_bf16.h>

#define Bb   128
#define Tt   2048
#define Ll   256
#define Vv   30
#define KVd  128
#define Hh   512
#define EMBd 256

#define K1 896    // EMB + KV + H
#define N1 2048   // 4H
#define KT1 28    // 896/32
#define K2 640    // H + KV
#define N2 512    // 4KV
#define KT2 20    // 640/32

typedef __bf16 bf16_t;
typedef bf16_t v16bf __attribute__((ext_vector_type(16)));
typedef float  v8f   __attribute__((ext_vector_type(8)));

__device__ __forceinline__ unsigned short f2bf(float f) {
    unsigned int u = __float_as_uint(f);
    u += 0x7fffu + ((u >> 16) & 1u);       // round-to-nearest-even
    return (unsigned short)(u >> 16);
}
__device__ __forceinline__ float bflo(unsigned int w) { return __uint_as_float(w << 16); }
__device__ __forceinline__ float bfhi(unsigned int w) { return __uint_as_float(w & 0xffff0000u); }
__device__ __forceinline__ float sigf(float x) { return 1.0f / (1.0f + expf(-x)); }

// ---------------- setup kernels ----------------

__global__ void cvt_bf16(const float* __restrict__ s, unsigned short* __restrict__ d, int n) {
    int i = blockIdx.x * blockDim.x + threadIdx.x;
    int stride = gridDim.x * blockDim.x;
    for (; i < n; i += stride) d[i] = f2bf(s[i]);
}

// pack W_ih1 (2048x384) ++ W_hh1 (2048x512) as B-fragments: [nt][kt] frags of 512 ushorts
__global__ void pack_W1(const float* __restrict__ Wih, const float* __restrict__ Whh,
                        unsigned short* __restrict__ dst) {
    int idx = blockIdx.x * blockDim.x + threadIdx.x;
    int total = N1 * K1;
    int stride = gridDim.x * blockDim.x;
    for (; idx < total; idx += stride) {
        int n = idx / K1, k = idx % K1;
        float w = (k < EMBd + KVd) ? Wih[n * (EMBd + KVd) + k] : Whh[n * Hh + (k - EMBd - KVd)];
        int kt = k >> 5;
        int e = (k & 7) + ((k & 16) >> 1);
        int lane = (n & 15) + ((k & 8) << 1);
        dst[(size_t)((n >> 4) * KT1 + kt) * 512 + lane * 16 + e] = f2bf(w);
    }
}

__global__ void pack_W2(const float* __restrict__ Wih, const float* __restrict__ Whh,
                        unsigned short* __restrict__ dst) {
    int idx = blockIdx.x * blockDim.x + threadIdx.x;
    int total = N2 * K2;
    int stride = gridDim.x * blockDim.x;
    for (; idx < total; idx += stride) {
        int n = idx / K2, k = idx % K2;
        float w = (k < Hh) ? Wih[n * Hh + k] : Whh[n * KVd + (k - Hh)];
        int kt = k >> 5;
        int e = (k & 7) + ((k & 16) >> 1);
        int lane = (n & 15) + ((k & 8) << 1);
        dst[(size_t)((n >> 4) * KT2 + kt) * 512 + lane * 16 + e] = f2bf(w);
    }
}

__global__ void init_state(const float* __restrict__ value,
                           float* __restrict__ h1, float* __restrict__ c1,
                           float* __restrict__ h2, float* __restrict__ c2,
                           float* __restrict__ ctx, unsigned short* __restrict__ X2f) {
    int idx = blockIdx.x * blockDim.x + threadIdx.x;   // 320*256 = 81920 exactly
    if (idx < Bb * Hh) { h1[idx] = 0.0f; c1[idx] = 0.0f; }
    if (idx < Bb * KVd) {
        h2[idx] = 0.0f; c2[idx] = 0.0f;
        int b = idx >> 7, d = idx & 127;
        ctx[idx] = value[(size_t)b * Tt * KVd + d];    // value[:,0,:]
    }
    if (idx < 8 * KT2 * 512) X2f[idx] = 0;
}

// ---------------- per-step kernels ----------------

// build A fragments of X1 = [emb | ctx | h1], 128 x 896
__global__ void prep_x1(const int* __restrict__ y, const float* __restrict__ E,
                        const float* __restrict__ ctx, const float* __restrict__ h1,
                        unsigned short* __restrict__ dst, int l) {
    int idx = blockIdx.x * blockDim.x + threadIdx.x;   // 448*256 = 128*896 exactly
    int b = idx / K1, k = idx % K1;
    float v;
    if (k < EMBd) {
        int tok = (l == 0) ? 0 : y[b * Ll + (l - 1)];
        v = E[tok * EMBd + k];
    } else if (k < EMBd + KVd) {
        v = ctx[b * KVd + (k - EMBd)];
    } else {
        v = h1[b * Hh + (k - EMBd - KVd)];
    }
    int kt = k >> 5;
    int e = (k & 7) + ((k & 16) >> 1);
    int lane = (b & 15) + ((k & 8) << 1);
    dst[(size_t)((b >> 4) * KT1 + kt) * 512 + lane * 16 + e] = f2bf(v);
}

// C(128 x ldc) = A(128 x 32*KT) * B(32*KT x ldc).
// One block = 256 threads = 8 waves; block computes a 128 x 64 macro-tile:
// wave w owns m-tile w and 4 n-tiles (register blocked; A-frag reused 4x,
// B-frags shared by all 8 waves through the WGP cache).
__global__ void wmma_gemm(const unsigned short* __restrict__ Af,
                          const unsigned short* __restrict__ Bf,
                          float* __restrict__ C, int KT, int ldc) {
    int mt   = threadIdx.x >> 5;      // wave id = m-tile
    int lane = threadIdx.x & 31;
    int nt0  = blockIdx.x * 4;

    const v16bf* ap  = (const v16bf*)(Af + (size_t)(mt * KT) * 512) + lane;
    const v16bf* bp0 = (const v16bf*)(Bf + (size_t)((nt0 + 0) * KT) * 512) + lane;
    const v16bf* bp1 = (const v16bf*)(Bf + (size_t)((nt0 + 1) * KT) * 512) + lane;
    const v16bf* bp2 = (const v16bf*)(Bf + (size_t)((nt0 + 2) * KT) * 512) + lane;
    const v16bf* bp3 = (const v16bf*)(Bf + (size_t)((nt0 + 3) * KT) * 512) + lane;

    v8f acc0 = {}, acc1 = {}, acc2 = {}, acc3 = {};
    for (int kt = 0; kt < KT; ++kt) {
        v16bf a  = ap[0];
        v16bf b0 = bp0[0];
        v16bf b1 = bp1[0];
        v16bf b2 = bp2[0];
        v16bf b3 = bp3[0];
        acc0 = __builtin_amdgcn_wmma_f32_16x16x32_bf16(false, a, false, b0, (short)0, acc0, false, false);
        acc1 = __builtin_amdgcn_wmma_f32_16x16x32_bf16(false, a, false, b1, (short)0, acc1, false, false);
        acc2 = __builtin_amdgcn_wmma_f32_16x16x32_bf16(false, a, false, b2, (short)0, acc2, false, false);
        acc3 = __builtin_amdgcn_wmma_f32_16x16x32_bf16(false, a, false, b3, (short)0, acc3, false, false);
        ap += 32; bp0 += 32; bp1 += 32; bp2 += 32; bp3 += 32;   // 512 ushorts = 1 frag
    }

    int col = lane & 15;
    int ro  = (lane >> 4) << 3;
    float* cp = C + (size_t)(mt * 16 + ro) * ldc + nt0 * 16 + col;
#pragma unroll
    for (int r = 0; r < 8; ++r) {
        cp[(size_t)r * ldc]      = acc0[r];
        cp[(size_t)r * ldc + 16] = acc1[r];
        cp[(size_t)r * ldc + 32] = acc2[r];
        cp[(size_t)r * ldc + 48] = acc3[r];
    }
}

// LSTM1 activation; also packs h1 (K-range [0,512)) into X2 A-fragments
__global__ void act1(const float* __restrict__ g, const float* __restrict__ bi,
                     const float* __restrict__ bh, float* __restrict__ c1,
                     float* __restrict__ h1, unsigned short* __restrict__ X2f) {
    int idx = blockIdx.x * blockDim.x + threadIdx.x;   // 256*256 = 128*512
    int b = idx >> 9, j = idx & 511;
    const float* gr = g + (size_t)b * N1;
    float ig = gr[j]        + bi[j]        + bh[j];
    float fg = gr[512 + j]  + bi[512 + j]  + bh[512 + j];
    float gg = gr[1024 + j] + bi[1024 + j] + bh[1024 + j];
    float og = gr[1536 + j] + bi[1536 + j] + bh[1536 + j];
    float c = sigf(fg) * c1[idx] + sigf(ig) * tanhf(gg);
    float h = sigf(og) * tanhf(c);
    c1[idx] = c; h1[idx] = h;
    int kt = j >> 5;
    int e = (j & 7) + ((j & 16) >> 1);
    int lane = (b & 15) + ((j & 8) << 1);
    X2f[(size_t)((b >> 4) * KT2 + kt) * 512 + lane * 16 + e] = f2bf(h);
}

// LSTM2 activation; packs h2 (K-range [512,640)) into X2 fragments for the NEXT step
__global__ void act2(const float* __restrict__ g, const float* __restrict__ bi,
                     const float* __restrict__ bh, float* __restrict__ c2,
                     float* __restrict__ h2, unsigned short* __restrict__ X2f) {
    int idx = blockIdx.x * blockDim.x + threadIdx.x;   // 64*256 = 128*128
    int b = idx >> 7, j = idx & 127;
    const float* gr = g + (size_t)b * N2;
    float ig = gr[j]       + bi[j]       + bh[j];
    float fg = gr[128 + j] + bi[128 + j] + bh[128 + j];
    float gg = gr[256 + j] + bi[256 + j] + bh[256 + j];
    float og = gr[384 + j] + bi[384 + j] + bh[384 + j];
    float c = sigf(fg) * c2[idx] + sigf(ig) * tanhf(gg);
    float h = sigf(og) * tanhf(c);
    c2[idx] = c; h2[idx] = h;
    int kt = 16 + (j >> 5);                 // k = 512 + j
    int e = (j & 7) + ((j & 16) >> 1);
    int lane = (b & 15) + ((j & 8) << 1);
    X2f[(size_t)((b >> 4) * KT2 + kt) * 512 + lane * 16 + e] = f2bf(h);
}

// fused attention: q-proj, energy over T (bf16), masked softmax, context, Wc proj
__global__ void attn_kernel(const unsigned short* __restrict__ encb,
                            const unsigned short* __restrict__ valb,
                            const int* __restrict__ enc_len,
                            const float* __restrict__ h2,
                            const float* __restrict__ Wq, const float* __restrict__ bq,
                            const float* __restrict__ Wc, const float* __restrict__ bc,
                            float* __restrict__ qout, float* __restrict__ ctxout,
                            float* __restrict__ attn_plot, int l) {
    __shared__ float sH[KVd], sQ[KVd], sE[Tt], sRed[256], sP[8 * KVd], sCtx[KVd];
    int b = blockIdx.x, tid = threadIdx.x;

    if (tid < KVd) sH[tid] = h2[b * KVd + tid];
    __syncthreads();
    if (tid < KVd) {
        float acc = bq[tid];
        for (int k = 0; k < KVd; ++k) acc += sH[k] * Wq[tid * KVd + k];
        sQ[tid] = acc;
        qout[b * KVd + tid] = acc;
    }
    __syncthreads();

    int elen = enc_len[b];
    float lm = -__builtin_inff();
    for (int t = tid; t < Tt; t += 256) {
        float acc = 0.0f;
        const uint4* row = (const uint4*)(encb + ((size_t)b * Tt + t) * KVd);
#pragma unroll 4
        for (int k = 0; k < KVd / 8; ++k) {      // 16 x b128 loads
            uint4 w = row[k];
            acc += bflo(w.x) * sQ[8 * k + 0] + bfhi(w.x) * sQ[8 * k + 1];
            acc += bflo(w.y) * sQ[8 * k + 2] + bfhi(w.y) * sQ[8 * k + 3];
            acc += bflo(w.z) * sQ[8 * k + 4] + bfhi(w.z) * sQ[8 * k + 5];
            acc += bflo(w.w) * sQ[8 * k + 6] + bfhi(w.w) * sQ[8 * k + 7];
        }
        float e = (t >= elen) ? -__builtin_inff() : acc;
        sE[t] = e;
        lm = fmaxf(lm, e);
    }
    sRed[tid] = lm;
    __syncthreads();
    for (int s = 128; s > 0; s >>= 1) {
        if (tid < s) sRed[tid] = fmaxf(sRed[tid], sRed[tid + s]);
        __syncthreads();
    }
    float mx = sRed[0];
    __syncthreads();

    float ls = 0.0f;
    for (int t = tid; t < Tt; t += 256) {
        float ex = expf(sE[t] - mx);
        sE[t] = ex;
        ls += ex;
    }
    sRed[tid] = ls;
    __syncthreads();
    for (int s = 128; s > 0; s >>= 1) {
        if (tid < s) sRed[tid] += sRed[tid + s];
        __syncthreads();
    }
    float inv = 1.0f / sRed[0];
    __syncthreads();
    for (int t = tid; t < Tt; t += 256) {
        float a = sE[t] * inv;
        sE[t] = a;
        if (b == 0) attn_plot[(size_t)l * Tt + t] = a;
    }
    __syncthreads();

    // context accumulation: 8 t-groups x 32 d-quads (uint2 = 4 bf16 per lane)
    int dq = (tid & 31) * 4;
    int tg = tid >> 5;
    float a0 = 0.0f, a1 = 0.0f, a2 = 0.0f, a3 = 0.0f;
    for (int t = tg; t < Tt; t += 8) {
        uint2 w = *(const uint2*)(valb + ((size_t)b * Tt + t) * KVd + dq);
        float at = sE[t];
        a0 += at * bflo(w.x);
        a1 += at * bfhi(w.x);
        a2 += at * bflo(w.y);
        a3 += at * bfhi(w.y);
    }
    sP[tg * KVd + dq + 0] = a0;
    sP[tg * KVd + dq + 1] = a1;
    sP[tg * KVd + dq + 2] = a2;
    sP[tg * KVd + dq + 3] = a3;
    __syncthreads();
    if (tid < KVd) {
        float c = 0.0f;
#pragma unroll
        for (int g = 0; g < 8; ++g) c += sP[g * KVd + tid];
        sCtx[tid] = c;
    }
    __syncthreads();
    if (tid < KVd) {
        float acc = bc[tid];
        for (int k = 0; k < KVd; ++k) acc += sCtx[k] * Wc[tid * KVd + k];
        ctxout[b * KVd + tid] = acc;
    }
}

__global__ void logits_kernel(const float* __restrict__ q, const float* __restrict__ ctx,
                              const float* __restrict__ E, const float* __restrict__ b_char,
                              float* __restrict__ out, int l) {
    __shared__ float xq[2 * KVd];
    int b = blockIdx.x, tid = threadIdx.x;   // blockDim = 64
    for (int i = tid; i < 2 * KVd; i += 64)
        xq[i] = (i < KVd) ? q[b * KVd + i] : ctx[b * KVd + (i - KVd)];
    __syncthreads();
    if (tid < Vv) {
        float acc = b_char[tid];
        for (int k = 0; k < 2 * KVd; ++k) acc += xq[k] * E[tid * 2 * KVd + k];
        out[((size_t)b * Ll + l) * Vv + tid] = acc;
    }
}

// ---------------- host ----------------

extern "C" void kernel_launch(void* const* d_in, const int* in_sizes, int n_in,
                              void* d_out, int out_size, void* d_ws, size_t ws_size,
                              hipStream_t stream) {
    const float* enc_key = (const float*)d_in[0];
    const float* value   = (const float*)d_in[1];
    const int*   enc_len = (const int*)d_in[2];
    const int*   y       = (const int*)d_in[3];
    const float* E       = (const float*)d_in[4];
    const float* W_ih1   = (const float*)d_in[5];
    const float* W_hh1   = (const float*)d_in[6];
    const float* b_ih1   = (const float*)d_in[7];
    const float* b_hh1   = (const float*)d_in[8];
    const float* W_ih2   = (const float*)d_in[9];
    const float* W_hh2   = (const float*)d_in[10];
    const float* b_ih2   = (const float*)d_in[11];
    const float* b_hh2   = (const float*)d_in[12];
    const float* Wq      = (const float*)d_in[13];
    const float* bq      = (const float*)d_in[14];
    const float* Wc      = (const float*)d_in[15];
    const float* bc      = (const float*)d_in[16];
    const float* b_char  = (const float*)d_in[17];

    float* preds = (float*)d_out;                          // (B, L, V)
    float* attn_plot = preds + (size_t)Bb * Ll * Vv;       // (L, T)

    char* ws = (char*)d_ws;
    size_t off = 0;
    auto alloc = [&](size_t bytes) -> char* {
        char* p = ws + off;
        off = (off + bytes + 255) & ~(size_t)255;
        return p;
    };
    unsigned short* encb   = (unsigned short*)alloc((size_t)Bb * Tt * KVd * 2);
    unsigned short* valb   = (unsigned short*)alloc((size_t)Bb * Tt * KVd * 2);
    unsigned short* W1f    = (unsigned short*)alloc((size_t)(N1 / 16) * KT1 * 512 * 2);
    unsigned short* W2f    = (unsigned short*)alloc((size_t)(N2 / 16) * KT2 * 512 * 2);
    unsigned short* X1f    = (unsigned short*)alloc((size_t)8 * KT1 * 512 * 2);
    unsigned short* X2f    = (unsigned short*)alloc((size_t)8 * KT2 * 512 * 2);
    float* gates1 = (float*)alloc((size_t)Bb * N1 * 4);
    float* gates2 = (float*)alloc((size_t)Bb * N2 * 4);
    float* h1  = (float*)alloc((size_t)Bb * Hh * 4);
    float* c1  = (float*)alloc((size_t)Bb * Hh * 4);
    float* h2  = (float*)alloc((size_t)Bb * KVd * 4);
    float* c2  = (float*)alloc((size_t)Bb * KVd * 4);
    float* ctx = (float*)alloc((size_t)Bb * KVd * 4);
    float* qws = (float*)alloc((size_t)Bb * KVd * 4);

    cvt_bf16<<<8192, 256, 0, stream>>>(enc_key, encb, Bb * Tt * KVd);
    cvt_bf16<<<8192, 256, 0, stream>>>(value, valb, Bb * Tt * KVd);
    pack_W1<<<(N1 * K1 + 255) / 256, 256, 0, stream>>>(W_ih1, W_hh1, W1f);
    pack_W2<<<(N2 * K2 + 255) / 256, 256, 0, stream>>>(W_ih2, W_hh2, W2f);
    init_state<<<320, 256, 0, stream>>>(value, h1, c1, h2, c2, ctx, X2f);

    for (int l = 0; l < Ll; ++l) {
        prep_x1<<<448, 256, 0, stream>>>(y, E, ctx, h1, X1f, l);
        wmma_gemm<<<32, 256, 0, stream>>>(X1f, W1f, gates1, KT1, N1);   // 128x2048
        act1<<<256, 256, 0, stream>>>(gates1, b_ih1, b_hh1, c1, h1, X2f);
        wmma_gemm<<<8, 256, 0, stream>>>(X2f, W2f, gates2, KT2, N2);    // 128x512
        act2<<<64, 256, 0, stream>>>(gates2, b_ih2, b_hh2, c2, h2, X2f);
        attn_kernel<<<Bb, 256, 0, stream>>>(encb, valb, enc_len, h2, Wq, bq, Wc, bc,
                                            qws, ctx, attn_plot, l);
        logits_kernel<<<Bb, 64, 0, stream>>>(qws, ctx, E, b_char, preds, l);
    }
}
